// VmfVectorQuantizer_12163347382690
// MI455X (gfx1250) — matrix operand order repaired and checked by reading, not compiled, not run on hardware
//
#include <hip/hip_runtime.h>
#include <hip/hip_bf16.h>

// ---------------------------------------------------------------------------
// vMF vector quantizer forward, fused for gfx1250 (MI455X).
//   bs=32, dim_z=64, W=H=32  -> N = 32768 tokens, K = 1024 codes.
// Matrix work runs on v_wmma_f32_16x16x32_f16 (wave32 WMMA).
// ---------------------------------------------------------------------------

typedef __attribute__((ext_vector_type(16))) _Float16 v16h;
typedef __attribute__((ext_vector_type(8)))  float    v8f;

#define N_TOK   32768
#define KCODES  1024
#define DIM     64
#define TOK     16          // tokens per workgroup (one WMMA M-tile)
#define NTHR    128         // 4 waves of 32

// workspace byte offsets
#define WS_CBG1   0u                       // f16[1024*64] swizzled for GEMM1 B
#define WS_CBG2   131072u                  // f16[1024*64] swizzled for GEMM2 B
#define WS_GAVG   262144u                  // f32[1024] sum of probabilities
#define WS_GSCAL  266240u                  // f32[2]: {sum p*logp, sum z*(z-zq)*kappa}

// ---------------------------------------------------------------------------
// JAX-exact gumbel noise: threefry2x32 with key (0, 42), count = 2^25,
// iota split into halves (x0 = i, x1 = i + 2^24), then
// u = bits>>9 | 0x3f800000 (as float) - 1, scaled into [tiny, 1), g = -log(-log u)
// ---------------------------------------------------------------------------
__device__ __forceinline__ unsigned rotl32(unsigned x, int n) {
  return (x << n) | (x >> (32 - n));
}

__device__ __forceinline__ float gumbel_jax(unsigned i) {
  const unsigned half = 1u << 24;                 // count/2, count = 32768*1024
  const bool second = (i >= half);
  unsigned x0 = second ? (i - half) : i;
  unsigned x1 = second ? i          : (i + half);
  const unsigned ks0 = 0u, ks1 = 42u, ks2 = 0u ^ 42u ^ 0x1BD11BDAu;
  x0 += ks0; x1 += ks1;
#define TF_ROUND(r) { x0 += x1; x1 = rotl32(x1, r); x1 ^= x0; }
#define TF_FOUR_A  TF_ROUND(13) TF_ROUND(15) TF_ROUND(26) TF_ROUND(6)
#define TF_FOUR_B  TF_ROUND(17) TF_ROUND(29) TF_ROUND(16) TF_ROUND(24)
  TF_FOUR_A; x0 += ks1; x1 += ks2 + 1u;
  TF_FOUR_B; x0 += ks2; x1 += ks0 + 2u;
  TF_FOUR_A; x0 += ks0; x1 += ks1 + 3u;
  TF_FOUR_B; x0 += ks1; x1 += ks2 + 4u;
  TF_FOUR_A; x0 += ks2; x1 += ks0 + 5u;
#undef TF_FOUR_B
#undef TF_FOUR_A
#undef TF_ROUND
  unsigned bits = second ? x1 : x0;
  float f = __uint_as_float((bits >> 9) | 0x3f800000u) - 1.0f;   // [0,1)
  const float tiny = 1.1754944e-38f;
  float u = fmaxf(f * (1.0f - tiny) + tiny, tiny);
  return -__logf(-__logf(u));
}

__device__ __forceinline__ v8f wmma_f16(v16h a, v16h b, v8f c) {
  return __builtin_amdgcn_wmma_f32_16x16x32_f16(false, a, false, b, (short)0, c,
                                                false, false);
}

// ---------------------------------------------------------------------------
// Kernel: zero the global accumulators (graph-safe re-init each launch)
// ---------------------------------------------------------------------------
__global__ void vq_zero(float* __restrict__ gavg, float* __restrict__ gscal) {
  int i = threadIdx.x;
  for (int c = i; c < KCODES; c += 256) gavg[c] = 0.0f;
  if (i < 2) gscal[i] = 0.0f;
}

// ---------------------------------------------------------------------------
// Kernel: normalize codebook rows, write f16 in WMMA-B-fragment-swizzled
// layouts for both GEMMs so the main kernel loads B as contiguous b128s.
//   GEMM1 B tile (k=d in [s*32,s*32+32), n=code): g1[((gct*2+s)*32 + ((k>>3)&1)*16 + j)*16
//          + (k&7) + (k>>4)*8] = cbn[gct*16 + j][s*32 + k]
//   GEMM2 B tile (k=code in [kt*32,..), n=d):     g2[((kt*4+dt)*32 + ((k>>3)&1)*16 + j)*16
//          + (k&7) + (k>>4)*8] = cbn[kt*32 + k][dt*16 + j]
// ---------------------------------------------------------------------------
__global__ void vq_prep_cb(const float* __restrict__ cb,
                           _Float16* __restrict__ g1,
                           _Float16* __restrict__ g2) {
  int c = blockIdx.x * blockDim.x + threadIdx.x;   // code index 0..1023
  if (c >= KCODES) return;
  float v[DIM];
  float ss = 0.0f;
#pragma unroll
  for (int d = 0; d < DIM; ++d) { v[d] = cb[c * DIM + d]; ss += v[d] * v[d]; }
  float inv = rsqrtf(ss);
  const int gct = c >> 4, j = c & 15;    // GEMM1 code-tile / column
  const int kt = c >> 5, lk = c & 31;    // GEMM2 k-tile / lane
#pragma unroll
  for (int d = 0; d < DIM; ++d) {
    _Float16 h = (_Float16)(v[d] * inv);
    int s = d >> 5, l = d & 31;
    g1[((gct * 2 + s) * 32 + ((l >> 3) & 1) * 16 + j) * 16 + (l & 7) + (l >> 4) * 8] = h;
    int dt = d >> 4, jj = d & 15;
    g2[((kt * 4 + dt) * 32 + ((lk >> 3) & 1) * 16 + jj) * 16 + (lk & 7) + (lk >> 4) * 8] = h;
  }
}

// ---------------------------------------------------------------------------
// Main fused kernel: 1 workgroup (4 waves) per 16-token tile.
// ---------------------------------------------------------------------------
__global__ void __launch_bounds__(NTHR)
vq_main(const float* __restrict__ z, const _Float16* __restrict__ cbG1,
        const _Float16* __restrict__ cbG2, const float* __restrict__ kappa_p,
        float* __restrict__ zout, float* __restrict__ gavg,
        float* __restrict__ gscal) {
  __shared__ _Float16 logh[TOK * KCODES];   // logits, later encodings (f16) 32KB
  __shared__ float    zs[DIM * TOK];        // staged z tile, [d*16+m]        4KB
  __shared__ float    redA[NTHR], redB[NTHR], redC[NTHR], redD[NTHR];
  __shared__ float    avgp[KCODES];         // per-WG prob sums               4KB
  __shared__ float    kred[NTHR];

  const int tid = threadIdx.x;
  const int t0  = blockIdx.x * TOK;      // first token of tile
  const int bb  = t0 >> 10;              // batch index (1024 tokens / image)
  const int pos = t0 & 1023;             // w*32+h position within image
  const float kappa = kappa_p[0];
  const float inv_tau = 1.0f;            // TAU = 1.0

  // stage z tile (coalesced: 16 consecutive h per d) and clear avgp
  for (int e = tid; e < DIM * TOK; e += NTHR) {
    int m = e & 15, d = e >> 4;
    zs[d * TOK + m] = z[(size_t)bb * 65536 + (size_t)d * 1024 + pos + m];
  }
  for (int c = tid; c < KCODES; c += NTHR) avgp[c] = 0.0f;
  __syncthreads();

  const int wv = tid >> 5;               // wave 0..3
  const int l  = tid & 31;               // lane
  const int m_a = l & 15, hi = l >> 4;

  // --- A fragments for GEMM1: z tile, 2 k-steps over D=64 (f16) -----------
  // 16-bit A 16x32 layout: lane(m = l&15, hi = l>>4), VGPR v holds
  // K = (v>>2)*16 + hi*8 + (v&3)*2 (+1 in high half).
  v16h a0, a1;
#pragma unroll
  for (int v = 0; v < 8; ++v) {
    int d0 = (v >> 2) * 16 + hi * 8 + (v & 3) * 2;
    a0[2 * v]     = (_Float16)zs[(d0)*TOK + m_a];
    a0[2 * v + 1] = (_Float16)zs[(d0 + 1) * TOK + m_a];
    a1[2 * v]     = (_Float16)zs[(32 + d0) * TOK + m_a];
    a1[2 * v + 1] = (_Float16)zs[(32 + d0 + 1) * TOK + m_a];
  }

  // --- GEMM1: wave wv -> codes [wv*256, wv*256+256), 16 tiles x 2 k-steps --
  for (int ct = 0; ct < 16; ++ct) {
    const int gct = wv * 16 + ct;
    const v16h bf0 = *(const v16h*)(cbG1 + ((gct * 2 + 0) * 32 + l) * 16);
    const v16h bf1 = *(const v16h*)(cbG1 + ((gct * 2 + 1) * 32 + l) * 16);
    if (ct + 1 < 16)
      __builtin_prefetch((const void*)(cbG1 + (((gct + 1) * 2) * 32 + l) * 16), 0, 0);
    v8f c = {};
    c = wmma_f16(a0, bf0, c);
    c = wmma_f16(a1, bf1, c);
    // D layout: VGPR r, lanes 0-15 -> (M=r, N=l), lanes 16-31 -> (M=r+8, N=l-16)
    const int code0 = gct * 16 + m_a;
#pragma unroll
    for (int r = 0; r < 8; ++r)
      logh[(r + hi * 8) * KCODES + code0] = (_Float16)(kappa * c[r]);
  }
  __syncthreads();

  // --- softmax + gumbel-softmax (online, 8 threads per token row) ----------
  const int mrow = tid >> 3, sub = tid & 7, c0 = sub * 128;
  const unsigned gbase = (unsigned)(t0 + mrow) * 1024u;
  float m1 = -3.4e38f, s1 = 0.0f, m2 = -3.4e38f, s2 = 0.0f;
  for (int t = 0; t < 128; ++t) {
    const int c = c0 + t;
    const float L = (float)logh[mrow * KCODES + c];
    const float g = gumbel_jax(gbase + (unsigned)c);
    float x = L;
    if (x > m1) { s1 = s1 * __expf(m1 - x) + 1.0f; m1 = x; }
    else        { s1 += __expf(x - m1); }
    x = (L + g) * inv_tau;
    if (x > m2) { s2 = s2 * __expf(m2 - x) + 1.0f; m2 = x; }
    else        { s2 += __expf(x - m2); }
  }
  redA[mrow * 8 + sub] = m1; redB[mrow * 8 + sub] = s1;
  redC[mrow * 8 + sub] = m2; redD[mrow * 8 + sub] = s2;
  __syncthreads();
  float M1 = -3.4e38f, M2 = -3.4e38f, S1 = 0.0f, S2 = 0.0f;
#pragma unroll
  for (int j = 0; j < 8; ++j) {
    M1 = fmaxf(M1, redA[mrow * 8 + j]);
    M2 = fmaxf(M2, redC[mrow * 8 + j]);
  }
#pragma unroll
  for (int j = 0; j < 8; ++j) {
    S1 += redB[mrow * 8 + j] * __expf(redA[mrow * 8 + j] - M1);
    S2 += redD[mrow * 8 + j] * __expf(redC[mrow * 8 + j] - M2);
  }
  const float invS1 = 1.0f / S1, invS2 = 1.0f / S2, lgS1 = __logf(S1);

  float kldd = 0.0f;
  for (int t = 0; t < 128; ++t) {
    const int c = c0 + t;
    const float L = (float)logh[mrow * KCODES + c];
    const float g = gumbel_jax(gbase + (unsigned)c);
    const float p = __expf(L - M1) * invS1;       // softmax prob
    kldd += p * ((L - M1) - lgS1);                // p * log p
    atomicAdd(&avgp[c], p);                       // ds_add_f32
    const float e = __expf((L + g) * inv_tau - M2) * invS2;  // gumbel-softmax
    logh[mrow * KCODES + c] = (_Float16)e;        // overwrite with encodings
  }
  kred[tid] = kldd;
  __syncthreads();
  for (int off = NTHR / 2; off > 0; off >>= 1) {
    if (tid < off) kred[tid] += kred[tid + off];
    __syncthreads();
  }
  if (tid == 0) atomicAdd(&gscal[0], kred[0]);
  for (int c = tid; c < KCODES; c += NTHR) atomicAdd(&gavg[c], avgp[c]);
  __syncthreads();

  // --- GEMM2: z_q[16 x 64] = enc[16 x 1024] @ cb[1024 x 64] ----------------
  // wave wv owns output channels [wv*16, wv*16+16); 32 k-steps of 32.
  v8f acc = {};
  for (int kt = 0; kt < 32; ++kt) {
    v16h Af;
#pragma unroll
    for (int v = 0; v < 8; ++v) {
      int kk = kt * 32 + (v >> 2) * 16 + hi * 8 + (v & 3) * 2;
      Af[2 * v]     = logh[m_a * KCODES + kk];
      Af[2 * v + 1] = logh[m_a * KCODES + kk + 1];
    }
    const v16h Bf = *(const v16h*)(cbG2 + ((kt * 4 + wv) * 32 + l) * 16);
    if (kt + 1 < 32)
      __builtin_prefetch((const void*)(cbG2 + (((kt + 1) * 4 + wv) * 32 + l) * 16), 0, 0);
    acc = wmma_f16(Af, Bf, acc);
  }

  // store z_to_decoder [b, d, w, h] and fuse continuous-KLD partial
  const int d = wv * 16 + m_a;
  float kc = 0.0f;
#pragma unroll
  for (int r = 0; r < 8; ++r) {
    const int mm = r + hi * 8;
    const float zq = acc[r];
    const float zv = zs[d * TOK + mm];
    kc += zv * (zv - zq);
    zout[(size_t)bb * 65536 + (size_t)d * 1024 + pos + mm] = zq;
  }
  kc *= kappa;
  kred[tid] = kc;
  __syncthreads();
  for (int off = NTHR / 2; off > 0; off >>= 1) {
    if (tid < off) kred[tid] += kred[tid + off];
    __syncthreads();
  }
  if (tid == 0) atomicAdd(&gscal[1], kred[0]);
}

// ---------------------------------------------------------------------------
// Kernel: finalize loss + perplexity into d_out tail
// ---------------------------------------------------------------------------
__global__ void vq_finalize(const float* __restrict__ gavg,
                            const float* __restrict__ gscal,
                            float* __restrict__ out) {
  __shared__ float r[256];
  int tid = threadIdx.x;
  float part = 0.0f;
  for (int c = tid; c < KCODES; c += 256) {
    float t = gavg[c] * (1.0f / (float)N_TOK);    // avg_probs
    part += t * __logf(t + 1e-7f);
  }
  r[tid] = part;
  __syncthreads();
  for (int off = 128; off > 0; off >>= 1) {
    if (tid < off) r[tid] += r[tid + off];
    __syncthreads();
  }
  if (tid == 0) {
    // loss = kld_discrete + kld_continuous, both are (total sums)/bs
    out[2097152] = (gscal[0] + gscal[1]) * (1.0f / 32.0f);
    out[2097153] = __expf(-r[0]);
  }
}

// ---------------------------------------------------------------------------
extern "C" void kernel_launch(void* const* d_in, const int* in_sizes, int n_in,
                              void* d_out, int out_size, void* d_ws, size_t ws_size,
                              hipStream_t stream) {
  const float* z     = (const float*)d_in[0];   // (32, 64, 32, 32)
  const float* cb    = (const float*)d_in[1];   // (1024, 64)
  const float* kappa = (const float*)d_in[2];   // scalar
  float* out = (float*)d_out;                   // 2097152 + 2 floats

  char* ws = (char*)d_ws;
  _Float16* cbG1 = (_Float16*)(ws + WS_CBG1);
  _Float16* cbG2 = (_Float16*)(ws + WS_CBG2);
  float*    gavg = (float*)(ws + WS_GAVG);
  float*    gscal = (float*)(ws + WS_GSCAL);

  vq_zero<<<1, 256, 0, stream>>>(gavg, gscal);
  vq_prep_cb<<<4, 256, 0, stream>>>(cb, cbG1, cbG2);
  vq_main<<<N_TOK / TOK, NTHR, 0, stream>>>(z, cbG1, cbG2, kappa, out, gavg,
                                            gscal);
  vq_finalize<<<1, 256, 0, stream>>>(gavg, gscal, out);
}
